// PerformerAttention_28862180229423
// MI455X (gfx1250) — matrix operand order repaired and checked by reading, not compiled, not run on hardware
//
#include <hip/hip_runtime.h>

// Performer attention forward for MI455X (gfx1250, wave32, WMMA, async-LDS,
// double-buffered GEMM tiles). proj (d_in[4]) is dead code in the reference.

typedef __attribute__((ext_vector_type(16))) __bf16 v16bf;
typedef __attribute__((ext_vector_type(8)))  float  v8f;

#define B_   2
#define N_   4096
#define DIM_ 1024
#define H_   16
#define DH_  64
#define M_TOTAL (B_ * N_)   // 8192
#define NCHUNK 8            // n-split for context partials

__device__ __forceinline__ float  bf2f(__bf16 v) { return (float)v; }
__device__ __forceinline__ __bf16 f2bf(float v)  { return (__bf16)v; }

// 16-byte global -> LDS copy. Prefer the gfx1250 async-to-LDS path
// (GLOBAL_LOAD_ASYNC_TO_LDS_B128, tracked by ASYNCcnt); fall back to a
// synchronous VGPR round-trip if the builtin is unavailable.
#if defined(__gfx1250__) && __has_builtin(__builtin_amdgcn_global_load_async_to_lds_b128)
#define HAS_ASYNC_LDS 1
#else
#define HAS_ASYNC_LDS 0
#endif

typedef int v4i32 __attribute__((vector_size(16)));
typedef __attribute__((address_space(1))) v4i32* gptr_t;   // global int4*
typedef __attribute__((address_space(3))) v4i32* lptr_t;   // LDS int4*

__device__ __forceinline__ void cp16(void* l, const void* g) {
#if HAS_ASYNC_LDS
  __builtin_amdgcn_global_load_async_to_lds_b128((gptr_t)g, (lptr_t)l, 0, 0);
#else
  *(uint4*)l = *(const uint4*)g;
#endif
}

__device__ __forceinline__ void async_wait_all() {
#if HAS_ASYNC_LDS
#if __has_builtin(__builtin_amdgcn_s_wait_asynccnt)
  __builtin_amdgcn_s_wait_asynccnt(0);
#else
  asm volatile("s_wait_asynccnt 0x0" ::: "memory");
#endif
#endif
}

// ---- WMMA fragment loaders (ISA 7.12.2, 16-bit layouts, wave32) ----
// A: 16x32 (MxK). LDS row-major [M][kstride]. lane%16 = M row, lane/16 = h.
// Per lane: K = {8h..8h+7} then {16+8h..16+8h+7} -> two contiguous 16B reads.
__device__ __forceinline__ v16bf load_frag_a(const __bf16* lds, int mbase,
                                             int kbase, int kstride) {
  int lane = threadIdx.x & 31;
  int h = lane >> 4;
  const __bf16* p = lds + (size_t)(mbase + (lane & 15)) * kstride + kbase;
  v16bf a;
  *((uint4*)&a)     = *(const uint4*)(p + (h << 3));
  *((uint4*)&a + 1) = *(const uint4*)(p + 16 + (h << 3));
  return a;
}

// B: 32x16 (KxN). LDS column-major [N][kstride] (lds[col*kstride + k]).
// lane%16 = N col, per lane: K = {8h..8h+7} then {16+8h..16+8h+7}.
__device__ __forceinline__ v16bf load_frag_b(const __bf16* lds, int nbase,
                                             int kbase, int kstride) {
  int lane = threadIdx.x & 31;
  int h = lane >> 4;
  const __bf16* p = lds + (size_t)(nbase + (lane & 15)) * kstride + kbase;
  v16bf b;
  *((uint4*)&b)     = *(const uint4*)(p + (h << 3));
  *((uint4*)&b + 1) = *(const uint4*)(p + 16 + (h << 3));
  return b;
}

#define WMMA_BF16(a, b, c) \
  __builtin_amdgcn_wmma_f32_16x16x32_bf16(false, (a), false, (b), (short)0, (c), false, false)

// ---- fp32 -> bf16 conversion (grid-stride) ----
__global__ void conv_f32_bf16_kernel(const float* __restrict__ src,
                                     __bf16* __restrict__ dst, int count) {
  for (int i = blockIdx.x * blockDim.x + threadIdx.x; i < count;
       i += gridDim.x * blockDim.x)
    dst[i] = f2bf(src[i]);
}

// ---- fp32 [K][Ncols] -> bf16 transposed [Ncols][K] (output-contiguous) ----
__global__ void conv_t_bf16_kernel(const float* __restrict__ src,
                                   __bf16* __restrict__ dst, int K, int Ncols) {
  int total = K * Ncols;
  for (int i = blockIdx.x * blockDim.x + threadIdx.x; i < total;
       i += gridDim.x * blockDim.x) {
    int n = i / K, k = i - n * K;
    dst[i] = f2bf(src[(size_t)k * Ncols + n]);
  }
}

// ---- GEMM1: qkv = x @ W_qkv, scattered into per-head q/k/v (bf16) ----
// M=8192, N=3072, K=1024. BM=BN=128, BK=64, double-buffered async tiles.
// Wt is pre-transposed [N][K]. 256 threads = 8 waves (2x4).
__global__ __launch_bounds__(256) void gemm_qkv_kernel(
    const __bf16* __restrict__ xb, const __bf16* __restrict__ wt,
    __bf16* __restrict__ qh, __bf16* __restrict__ kh, __bf16* __restrict__ vh) {
  __shared__ __bf16 ldsA[2][128 * 64];   // row-major [m][k]
  __shared__ __bf16 ldsB[2][128 * 64];   // col-major [n][k]
  const int n0 = blockIdx.x * 128;
  const int m0 = blockIdx.y * 128;
  const int t = threadIdx.x;
  const int lane = t & 31;
  const int wave = t >> 5;
  const int wm = wave >> 2;   // 0..1 -> 64 rows
  const int wn = wave & 3;    // 0..3 -> 32 cols
  v8f vzero = {};
  v8f acc[4][2];
#pragma unroll
  for (int mt = 0; mt < 4; ++mt)
#pragma unroll
    for (int nt = 0; nt < 2; ++nt) acc[mt][nt] = vzero;

  auto issue_tile = [&](int k0, int buf) {
#pragma unroll
    for (int i = 0; i < 4; ++i) {
      int idx = t + i * 256;            // 0..1023
      int row = idx >> 3;               // 0..127
      int seg = (idx & 7) * 8;          // bf16 offset, 16B granules
      cp16(&ldsA[buf][row * 64 + seg], xb + (size_t)(m0 + row) * DIM_ + k0 + seg);
      cp16(&ldsB[buf][row * 64 + seg], wt + (size_t)(n0 + row) * DIM_ + k0 + seg);
    }
  };

  issue_tile(0, 0);
  int cur = 0;
  for (int k0 = 0; k0 < DIM_; k0 += 64) {
    async_wait_all();       // tile `cur` resident
    __syncthreads();        // ...and tile `cur^1` fully consumed by all waves
    if (k0 + 64 < DIM_) issue_tile(k0 + 64, cur ^ 1);   // overlap with compute
#pragma unroll
    for (int kk = 0; kk < 64; kk += 32) {
      v16bf b0 = load_frag_b(ldsB[cur], wn * 32, kk, 64);
      v16bf b1 = load_frag_b(ldsB[cur], wn * 32 + 16, kk, 64);
#pragma unroll
      for (int mt = 0; mt < 4; ++mt) {
        v16bf a = load_frag_a(ldsA[cur], wm * 64 + mt * 16, kk, 64);
        acc[mt][0] = WMMA_BF16(a, b0, acc[mt][0]);
        acc[mt][1] = WMMA_BF16(a, b1, acc[mt][1]);
      }
    }
    cur ^= 1;
  }
  // Epilogue: scatter into q/k/v head layout [b][h][n][dh]
  const int hh = lane >> 4;
#pragma unroll
  for (int mt = 0; mt < 4; ++mt)
#pragma unroll
    for (int nt = 0; nt < 2; ++nt)
#pragma unroll
      for (int j = 0; j < 8; ++j) {
        int gr = m0 + wm * 64 + mt * 16 + j + (hh << 3);
        int gc = n0 + wn * 32 + nt * 16 + (lane & 15);
        int which = gc >> 10;
        int col = gc & 1023;
        int h = col >> 6, dh = col & 63;
        int bb = gr >> 12, nn = gr & 4095;
        size_t dst = (((size_t)bb * H_ + h) * N_ + nn) * DH_ + dh;
        __bf16 val = f2bf(acc[mt][nt][j]);
        if (which == 0)      qh[dst] = val;
        else if (which == 1) kh[dst] = val;
        else                 vh[dst] = val;
      }
}

// ---- context partials: part[c,bh,d,e] = sum_{n in chunk} k[n,d]*v[n,e] ----
// Also accumulates ksum partials (fused). grid (B*H, NCHUNK), 512 thr.
__global__ __launch_bounds__(512) void ctx_partial_kernel(
    const __bf16* __restrict__ kh, const __bf16* __restrict__ vh,
    float* __restrict__ part, float* __restrict__ kpart) {
  __shared__ __bf16 kT[64 * 32];   // [d][n] (A row-major, M=d, K=n)
  __shared__ __bf16 vT[64 * 32];   // [e][n] (B col-major, N=e, K=n)
  __shared__ float kred[64 * 8];
  const int bh = blockIdx.x;
  const int chunk = blockIdx.y;
  const __bf16* kbase = kh + (size_t)bh * N_ * DH_;
  const __bf16* vbase = vh + (size_t)bh * N_ * DH_;
  const int t = threadIdx.x;
  const int lane = t & 31;
  const int wave = t >> 5;          // 0..15
  const int wtm = wave >> 2, wtn = wave & 3;
  const int kd = t >> 3;            // 0..63
  const int kn = (t & 7) * 4;
  float ksacc = 0.f;
  v8f acc = {};
  const int nstart = chunk * (N_ / NCHUNK);
  for (int nn0 = nstart; nn0 < nstart + N_ / NCHUNK; nn0 += 32) {
    __syncthreads();
    {
      int nn = t >> 4;              // 0..31
      int d0 = (t & 15) * 4;
      const __bf16* kr = kbase + (size_t)(nn0 + nn) * DH_ + d0;
      const __bf16* vr = vbase + (size_t)(nn0 + nn) * DH_ + d0;
#pragma unroll
      for (int d = 0; d < 4; ++d) {
        kT[(d0 + d) * 32 + nn] = kr[d];
        vT[(d0 + d) * 32 + nn] = vr[d];
      }
    }
    __syncthreads();
    v16bf a = load_frag_a(kT, wtm * 16, 0, 32);
    v16bf b = load_frag_b(vT, wtn * 16, 0, 32);
    acc = WMMA_BF16(a, b, acc);
#pragma unroll
    for (int i = 0; i < 4; ++i) ksacc += bf2f(kT[kd * 32 + kn + i]);
  }
  kred[kd * 8 + (t & 7)] = ksacc;
  __syncthreads();
  if (t < 64) {
    float s = 0.f;
#pragma unroll
    for (int i = 0; i < 8; ++i) s += kred[t * 8 + i];
    kpart[(size_t)(chunk * 32 + bh) * 64 + t] = s;
  }
  const int hh = lane >> 4;
  float* pb = part + (size_t)(chunk * 32 + bh) * 4096;
#pragma unroll
  for (int j = 0; j < 8; ++j) {
    int d = wtm * 16 + j + (hh << 3);
    int e = wtn * 16 + (lane & 15);
    pb[d * 64 + e] = acc[j];
  }
}

// ---- reduce partials -> ctx^T bf16 [bh][e][d] + ksum fp32 ----
__global__ __launch_bounds__(256) void ctx_reduce_kernel(
    const float* __restrict__ part, const float* __restrict__ kpart,
    __bf16* __restrict__ ctxTb, float* __restrict__ ksm) {
  const int bh = blockIdx.x;
  const int t = threadIdx.x;
#pragma unroll
  for (int i = 0; i < 16; ++i) {
    int idx = t + i * 256;          // 0..4095 over [e][d]
    int e = idx >> 6, d = idx & 63;
    float s = 0.f;
#pragma unroll
    for (int c = 0; c < NCHUNK; ++c)
      s += part[(size_t)(c * 32 + bh) * 4096 + d * 64 + e];
    ctxTb[(size_t)bh * 4096 + idx] = f2bf(s);
  }
  if (t < 64) {
    float s = 0.f;
#pragma unroll
    for (int c = 0; c < NCHUNK; ++c) s += kpart[(size_t)(c * 32 + bh) * 64 + t];
    ksm[bh * 64 + t] = s;
  }
}

// ---- out[b,h,n,e] = (q @ ctx) * Dinv[n], merged into [b][n][h*64+e] bf16 ----
__global__ __launch_bounds__(256) void qctx_kernel(
    const __bf16* __restrict__ qh, const __bf16* __restrict__ ctxTb,
    const float* __restrict__ ksum, __bf16* __restrict__ attn) {
  __shared__ __bf16 qs[128 * 64];   // [n][d] row-major
  __shared__ __bf16 cT[64 * 64];    // [e][d] col-major for B
  __shared__ float  ks[64];
  __shared__ float  dinv[128];
  const int bh = blockIdx.y;
  const int b = bh >> 4, h = bh & 15;
  const int m0 = blockIdx.x * 128;
  const int t = threadIdx.x;
  const __bf16* qbase = qh + ((size_t)bh * N_ + m0) * DH_;
#pragma unroll
  for (int i = 0; i < 4; ++i) {
    int idx = t + i * 256;          // flat copy of 128x64 tile
    cp16(qs + idx * 8, qbase + (size_t)idx * 8);
  }
#pragma unroll
  for (int i = 0; i < 2; ++i) {
    int idx = t + i * 256;          // flat copy of 64x64 ctx^T
    cp16(cT + idx * 8, ctxTb + (size_t)bh * 4096 + idx * 8);
  }
  if (t < 64) ks[t] = ksum[bh * 64 + t];
  async_wait_all();
  __syncthreads();
  if (t < 128) {
    float s = 0.f;
#pragma unroll
    for (int d = 0; d < 64; ++d) s += bf2f(qs[t * 64 + d]) * ks[d];
    dinv[t] = 1.0f / s;
  }
  __syncthreads();
  const int lane = t & 31, wave = t >> 5;
  v8f vzero = {};
  v8f acc[4];
#pragma unroll
  for (int nt = 0; nt < 4; ++nt) acc[nt] = vzero;
#pragma unroll
  for (int k0 = 0; k0 < DH_; k0 += 32) {
    v16bf a = load_frag_a(qs, wave * 16, k0, 64);
#pragma unroll
    for (int nt = 0; nt < 4; ++nt) {
      v16bf bfr = load_frag_b(cT, nt * 16, k0, 64);
      acc[nt] = WMMA_BF16(a, bfr, acc[nt]);
    }
  }
  const int hh = lane >> 4;
#pragma unroll
  for (int nt = 0; nt < 4; ++nt)
#pragma unroll
    for (int j = 0; j < 8; ++j) {
      int rl = wave * 16 + j + (hh << 3);
      int e = nt * 16 + (lane & 15);
      float val = acc[nt][j] * dinv[rl];
      attn[((size_t)b * N_ + m0 + rl) * DIM_ + h * 64 + e] = f2bf(val);
    }
}

// ---- GEMM2: out = attn @ W_out + b_out (fp32 result). Wt [N][K]. ----
__global__ __launch_bounds__(256) void gemm_out_kernel(
    const __bf16* __restrict__ ab, const __bf16* __restrict__ wt,
    const float* __restrict__ bias, float* __restrict__ out) {
  __shared__ __bf16 ldsA[2][128 * 64];
  __shared__ __bf16 ldsB[2][128 * 64];
  const int n0 = blockIdx.x * 128;
  const int m0 = blockIdx.y * 128;
  const int t = threadIdx.x;
  const int lane = t & 31;
  const int wave = t >> 5;
  const int wm = wave >> 2;
  const int wn = wave & 3;
  v8f vzero = {};
  v8f acc[4][2];
#pragma unroll
  for (int mt = 0; mt < 4; ++mt)
#pragma unroll
    for (int nt = 0; nt < 2; ++nt) acc[mt][nt] = vzero;

  auto issue_tile = [&](int k0, int buf) {
#pragma unroll
    for (int i = 0; i < 4; ++i) {
      int idx = t + i * 256;
      int row = idx >> 3;
      int seg = (idx & 7) * 8;
      cp16(&ldsA[buf][row * 64 + seg], ab + (size_t)(m0 + row) * DIM_ + k0 + seg);
      cp16(&ldsB[buf][row * 64 + seg], wt + (size_t)(n0 + row) * DIM_ + k0 + seg);
    }
  };

  issue_tile(0, 0);
  int cur = 0;
  for (int k0 = 0; k0 < DIM_; k0 += 64) {
    async_wait_all();
    __syncthreads();
    if (k0 + 64 < DIM_) issue_tile(k0 + 64, cur ^ 1);
#pragma unroll
    for (int kk = 0; kk < 64; kk += 32) {
      v16bf b0 = load_frag_b(ldsB[cur], wn * 32, kk, 64);
      v16bf b1 = load_frag_b(ldsB[cur], wn * 32 + 16, kk, 64);
#pragma unroll
      for (int mt = 0; mt < 4; ++mt) {
        v16bf a = load_frag_a(ldsA[cur], wm * 64 + mt * 16, kk, 64);
        acc[mt][0] = WMMA_BF16(a, b0, acc[mt][0]);
        acc[mt][1] = WMMA_BF16(a, b1, acc[mt][1]);
      }
    }
    cur ^= 1;
  }
  const int hh = lane >> 4;
#pragma unroll
  for (int mt = 0; mt < 4; ++mt)
#pragma unroll
    for (int nt = 0; nt < 2; ++nt)
#pragma unroll
      for (int j = 0; j < 8; ++j) {
        int gr = m0 + wm * 64 + mt * 16 + j + (hh << 3);
        int gc = n0 + wn * 32 + nt * 16 + (lane & 15);
        out[(size_t)gr * DIM_ + gc] = acc[mt][nt][j] + bias[gc];
      }
}

extern "C" void kernel_launch(void* const* d_in, const int* in_sizes, int n_in,
                              void* d_out, int out_size, void* d_ws, size_t ws_size,
                              hipStream_t stream) {
  const float* x    = (const float*)d_in[0];
  const float* Wqkv = (const float*)d_in[1];
  const float* Wout = (const float*)d_in[2];
  const float* bout = (const float*)d_in[3];
  // d_in[4] (proj) is unused: q_prime/k_prime are dead code in the reference.

  char* ws = (char*)d_ws;
  size_t off = 0;
  auto alloc = [&](size_t bytes) -> void* {
    void* p = ws + off;
    off += (bytes + 255) & ~(size_t)255;
    return p;
  };
  __bf16* xb    = (__bf16*)alloc((size_t)M_TOTAL * DIM_ * 2);
  __bf16* wqkvt = (__bf16*)alloc((size_t)3 * DIM_ * DIM_ * 2);  // [3072][1024]
  __bf16* woutt = (__bf16*)alloc((size_t)DIM_ * DIM_ * 2);      // [1024][1024]
  __bf16* qh    = (__bf16*)alloc((size_t)B_ * H_ * N_ * DH_ * 2);
  __bf16* kh    = (__bf16*)alloc((size_t)B_ * H_ * N_ * DH_ * 2);
  __bf16* vh    = (__bf16*)alloc((size_t)B_ * H_ * N_ * DH_ * 2);
  float*  part  = (float*)alloc((size_t)NCHUNK * 32 * 4096 * 4);
  float*  kpart = (float*)alloc((size_t)NCHUNK * 32 * 64 * 4);
  __bf16* ctxTb = (__bf16*)alloc((size_t)32 * 4096 * 2);
  float*  ksm   = (float*)alloc((size_t)32 * 64 * 4);
  __bf16* attn  = (__bf16*)alloc((size_t)M_TOTAL * DIM_ * 2);
  (void)ws_size; (void)in_sizes; (void)n_in; (void)out_size;

  conv_f32_bf16_kernel<<<2048, 256, 0, stream>>>(x, xb, M_TOTAL * DIM_);
  conv_t_bf16_kernel<<<2048, 256, 0, stream>>>(Wqkv, wqkvt, DIM_, 3 * DIM_);
  conv_t_bf16_kernel<<<1024, 256, 0, stream>>>(Wout, woutt, DIM_, DIM_);

  gemm_qkv_kernel<<<dim3(3072 / 128, M_TOTAL / 128), 256, 0, stream>>>(
      xb, wqkvt, qh, kh, vh);

  ctx_partial_kernel<<<dim3(B_ * H_, NCHUNK), 512, 0, stream>>>(kh, vh, part, kpart);
  ctx_reduce_kernel<<<B_ * H_, 256, 0, stream>>>(part, kpart, ctxTb, ksm);

  qctx_kernel<<<dim3(N_ / 128, B_ * H_), 256, 0, stream>>>(qh, ctxTb, ksm, attn);

  gemm_out_kernel<<<dim3(DIM_ / 128, M_TOTAL / 128), 256, 0, stream>>>(
      attn, woutt, bout, (float*)d_out);
}
